// Attention_63282048139810
// MI455X (gfx1250) — matrix operand order repaired and checked by reading, not compiled, not run on hardware
//
#include <hip/hip_runtime.h>

typedef __attribute__((ext_vector_type(16))) _Float16 v16h;
typedef __attribute__((ext_vector_type(8)))  _Float16 h8;
typedef __attribute__((ext_vector_type(8)))  float    v8f;
typedef __attribute__((ext_vector_type(8)))  float    f8v;
typedef __attribute__((ext_vector_type(4)))  unsigned int u32x4;
typedef __attribute__((ext_vector_type(4)))  int      i32x4;
typedef __attribute__((ext_vector_type(8)))  int      i32x8;

#define BATCH 4
#define SEQ   2048
#define DIMM  512
#define NHEAD 8
#define DK    64
#define BN    (BATCH * SEQ)      // 8192 rows total

#if defined(__has_builtin)
#if __has_builtin(__builtin_amdgcn_tensor_load_to_lds) && \
    __has_builtin(__builtin_amdgcn_s_wait_tensorcnt)
#define HAVE_TDM 1
#endif
#endif

#ifdef HAVE_TDM
#warning "CDNA5: TDM tensor_load_to_lds path ENABLED"
#else
#warning "CDNA5: TDM builtin not found - cooperative LDS fallback in use"
#endif

static __device__ __forceinline__ v8f wmma32(v16h a, v16h b, v8f c) {
  // D = A(16x32 f16) * B(32x16 f16) + C(16x16 f32)
  return __builtin_amdgcn_wmma_f32_16x16x32_f16(false, a, false, b, (short)0, c,
                                                false, false);
}

static __device__ __forceinline__ v8f vzero8() {
  v8f z;
#pragma unroll
  for (int e = 0; e < 8; ++e) z[e] = 0.0f;
  return z;
}

// A fragment (16 rows x 32 K) from a row-major f16 matrix (global or LDS).
// K order used here (contiguous, B loads match): lane<16 holds K = kb+{0..15}; lane>=16 holds K = kb+{16..31}.
static __device__ __forceinline__ v16h load_a_f16(const _Float16* p, int ld,
                                                  int kb, int lane) {
  const int lr = lane & 15, half = lane >> 4;
  const _Float16* row = p + lr * ld;
  h8 lo = *(const h8*)(row + kb + half * 16);
  h8 hi = *(const h8*)(row + kb + half * 16 + 8);
  v16h a;
#pragma unroll
  for (int e = 0; e < 8; ++e) { a[e] = lo[e]; a[e + 8] = hi[e]; }
  return a;
}

#ifdef HAVE_TDM
// One TDM tensor_load_to_lds of a 2-D f16 tile (tile1 rows x tile0 elements,
// row stride stride0 elements; tile1==0 -> 1-D).  D# per CDNA5 ISA 8.3/8.4.
static __device__ __forceinline__ void tdm_load_f16(
    unsigned int lds_off, const _Float16* gptr,
    unsigned int tile0, unsigned int tile1, unsigned long long stride0) {
  const unsigned long long ga = (unsigned long long)(uintptr_t)gptr;
  const unsigned int td0 = 1u << 20, td1 = 1u << 20;  // huge -> no OOB clamp
  u32x4 g0;
  g0[0] = 1u;                                   // count=1, user mode
  g0[1] = lds_off;                              // lds_addr (bytes)
  g0[2] = (unsigned int)ga;                     // global_addr[31:0]
  g0[3] = (unsigned int)((ga >> 32) & 0x01FFFFFFu) | 0x80000000u;  // [56:32]|type=2
  i32x8 g1;
  g1[0] = 0x00010000;                           // wg_mask=0, data_size=1 (2B)
  g1[1] = (int)((td0 & 0xFFFFu) << 16);
  g1[2] = (int)(((td0 >> 16) & 0xFFFFu) | ((td1 & 0xFFFFu) << 16));
  g1[3] = (int)(((td1 >> 16) & 0xFFFFu) | (tile0 << 16));
  g1[4] = (int)(tile1 & 0xFFFFu);
  g1[5] = (int)(unsigned int)(stride0 & 0xFFFFFFFFull);
  g1[6] = (int)(unsigned int)((stride0 >> 32) & 0xFFFFull);
  g1[7] = 0;
  i32x4 z4;
#pragma unroll
  for (int e = 0; e < 4; ++e) z4[e] = 0;
#if __clang_major__ >= 23
  i32x8 z8;
#pragma unroll
  for (int e = 0; e < 8; ++e) z8[e] = 0;
  __builtin_amdgcn_tensor_load_to_lds(g0, g1, z4, z4, z8, 0);
#else
  __builtin_amdgcn_tensor_load_to_lds(g0, g1, z4, z4, 0);
#endif
}
#endif  // HAVE_TDM

// ---------------------------------------------------------------------------
// Pre-pass 1: x (f32) -> xh (f16).
// ---------------------------------------------------------------------------
__global__ __launch_bounds__(256) void prep_x(const float* __restrict__ x,
                                              _Float16* __restrict__ xh) {
  const size_t i = ((size_t)blockIdx.x * 256 + threadIdx.x) * 8;
  f8v v = *(const f8v*)(x + i);
  h8 o;
#pragma unroll
  for (int e = 0; e < 8; ++e) o[e] = (_Float16)v[e];
  *(h8*)(xh + i) = o;
}

// ---------------------------------------------------------------------------
// Pre-pass 2: W[k][c] (f32) -> Wt[c][k] (f16), 4 matrices (q,k,v,o).
// ---------------------------------------------------------------------------
__global__ __launch_bounds__(256) void prep_w(
    const float* __restrict__ Wq, const float* __restrict__ Wk,
    const float* __restrict__ Wv, const float* __restrict__ Wo,
    _Float16* __restrict__ Wt) {
  const int m = blockIdx.y;
  const float* W = (m == 0) ? Wq : (m == 1) ? Wk : (m == 2) ? Wv : Wo;
  _Float16* dst = Wt + (size_t)m * DIMM * DIMM;
  const int o = blockIdx.x * 256 + threadIdx.x;
  const int c = o >> 6;
  const int k0 = (o & 63) * 8;
  h8 st;
#pragma unroll
  for (int e = 0; e < 8; ++e) st[e] = (_Float16)W[(size_t)(k0 + e) * DIMM + c];
  *(h8*)(dst + (size_t)c * DIMM + k0) = st;
}

#define LDB(wb0, t, kbv) (*(const v16h*)((wb0) + (size_t)(t) * 16 * DIMM + (kbv)))

// ---------------------------------------------------------------------------
// Kernel 1: QKV projections; rotation-free 2-stage pipeline.
// grid = (BN/16, 1, 3); each wave: 16x64 strip of X @ W.
// ---------------------------------------------------------------------------
__global__ __launch_bounds__(256) void proj_qkv(
    const _Float16* __restrict__ xh, const _Float16* __restrict__ Wt,
    const float* __restrict__ rcb,
    _Float16* __restrict__ qh, _Float16* __restrict__ kh,
    _Float16* __restrict__ vT) {
  const int lane = threadIdx.x & 31;
  const int wave = threadIdx.x >> 5;
  const int rt   = blockIdx.x;
  const int sel  = blockIdx.z;
  const _Float16* W = Wt + (size_t)sel * DIMM * DIMM;
  const int lr = lane & 15, half = lane >> 4;
  const int row0 = rt * 16;
  const int cbase = wave * 64;

  v8f acc[4];
#pragma unroll
  for (int t = 0; t < 4; ++t) acc[t] = vzero8();

  const _Float16* arow = xh + (size_t)row0 * DIMM;
  const _Float16* wb0  = W + (size_t)(cbase + lr) * DIMM + half * 16;

  v16h a0 = load_a_f16(arow, DIMM, 0, lane);
  v16h a1 = load_a_f16(arow, DIMM, 32, lane);
  v16h b0[4], b1[4];
#pragma unroll
  for (int t = 0; t < 4; ++t) { b0[t] = LDB(wb0, t, 0); b1[t] = LDB(wb0, t, 32); }

  for (int kk = 0; kk < DIMM / 32; kk += 2) {
#pragma unroll
    for (int t = 0; t < 4; ++t) acc[t] = wmma32(a0, b0[t], acc[t]);
    if (kk + 2 < DIMM / 32) {               // refill set0 while set1 computes
      const int kn = (kk + 2) * 32;
      a0 = load_a_f16(arow, DIMM, kn, lane);
#pragma unroll
      for (int t = 0; t < 4; ++t) b0[t] = LDB(wb0, t, kn);
    }
#pragma unroll
    for (int t = 0; t < 4; ++t) acc[t] = wmma32(a1, b1[t], acc[t]);
    if (kk + 3 < DIMM / 32) {               // refill set1 while set0 computes
      const int kn = (kk + 3) * 32;
      a1 = load_a_f16(arow, DIMM, kn, lane);
#pragma unroll
      for (int t = 0; t < 4; ++t) b1[t] = LDB(wb0, t, kn);
    }
  }

  const int h = wave;
  if (sel == 0) {
#pragma unroll
    for (int t = 0; t < 4; ++t) {
      const int d = t * 16 + lr;
      const float bias = rcb[h * DK + d];
#pragma unroll
      for (int r = 0; r < 8; ++r) {
        const int row = row0 + half * 8 + r;
        const int bb = row >> 11, nn = row & (SEQ - 1);
        qh[(((size_t)(bb * NHEAD + h) * SEQ + nn) << 6) + d] =
            (_Float16)(acc[t][r] * 0.125f + bias);
      }
    }
  } else if (sel == 1) {
#pragma unroll
    for (int t = 0; t < 4; ++t) {
      const int d = t * 16 + lr;
#pragma unroll
      for (int r = 0; r < 8; ++r) {
        const int row = row0 + half * 8 + r;
        const int bb = row >> 11, nn = row & (SEQ - 1);
        kh[(((size_t)(bb * NHEAD + h) * SEQ + nn) << 6) + d] = (_Float16)acc[t][r];
      }
    }
  } else {
    const int row = row0 + half * 8;
    const int bb = row >> 11, nn = row & (SEQ - 1);
#pragma unroll
    for (int t = 0; t < 4; ++t) {
      const int d = t * 16 + lr;
      h8 st;
#pragma unroll
      for (int r = 0; r < 8; ++r) st[r] = (_Float16)acc[t][r];
      *(h8*)(vT + ((size_t)(bb * NHEAD + h) * DK + d) * SEQ + nn) = st;
    }
  }
}

// ---------------------------------------------------------------------------
// Kernel 2: flash attention, S^T formulation.  grid = (16, 32), block = 256.
// Double-buffered LDS tiles; one barrier per chunk.  TDM path: wave 0 issues
// tensor_load_to_lds for chunk ch+1 right after the barrier, so the DMA runs
// concurrently with chunk ch's WMMA + softmax; s_wait_tensorcnt(0) next iter.
// ---------------------------------------------------------------------------
__global__ __launch_bounds__(256) void attn_fa(
    const _Float16* __restrict__ qh, const _Float16* __restrict__ kh,
    const _Float16* __restrict__ vT, _Float16* __restrict__ midh) {
  __shared__ __align__(16) _Float16 lk[2][32 * DK];   // K rows j0..j0+31
  __shared__ __align__(16) _Float16 lv[2][DK * 32];   // V^T rows d, 32 cols

  const int tid  = threadIdx.x;
  const int lane = tid & 31;
  const int wave = tid >> 5;
  const int it   = blockIdx.x * 8 + wave;   // query tile 0..127
  const int bh   = blockIdx.y;              // 0..31
  const int lr = lane & 15, half = lane >> 4;
  const int i0 = it * 16;

  const _Float16* qb = qh + (size_t)bh * SEQ * DK;
  const _Float16* kb = kh + (size_t)bh * SEQ * DK;
  const _Float16* vb = vT + (size_t)bh * DK * SEQ;

  const _Float16* qrow = qb + (size_t)(i0 + lr) * DK;
  const v16h bq0 = *(const v16h*)(qrow + half * 16);        // d 0..31
  const v16h bq1 = *(const v16h*)(qrow + 32 + half * 16);   // d 32..63

  v8f o0 = vzero8(), o1 = vzero8(), o2 = vzero8(), o3 = vzero8();
  float m = -1e30f, l = 0.0f;

  auto stage = [&](int chn, int buf) {
    const int jn = chn * 32;
#ifdef HAVE_TDM
    if (wave == 0) {
      tdm_load_f16((unsigned int)(uintptr_t)&lk[buf][0], kb + (size_t)jn * DK,
                   32 * DK, 0, 32 * DK);          // 1-D 4KB
      tdm_load_f16((unsigned int)(uintptr_t)&lv[buf][0], vb + jn,
                   32, DK, SEQ);                  // 2-D: 64 rows x 32 elems
    }
#else
    ((h8*)&lk[buf][0])[tid] = ((const h8*)(kb + (size_t)jn * DK))[tid];
    const int d = tid >> 2, part = tid & 3;
    *(h8*)(&lv[buf][0] + d * 32 + part * 8) =
        *(const h8*)(vb + (size_t)d * SEQ + jn + part * 8);
#endif
  };

  stage(0, 0);
  for (int ch = 0; ch < SEQ / 32; ++ch) {
    const int c = ch & 1;
#ifdef HAVE_TDM
    if (wave == 0) __builtin_amdgcn_s_wait_tensorcnt(0);
#endif
    __syncthreads();                      // buffer c is now valid for all waves
    if (ch + 1 < SEQ / 32) stage(ch + 1, c ^ 1);
    if (wave == 1 && ch + 2 < SEQ / 32) { // L2 prefetch two chunks ahead
      const int jn = (ch + 2) * 32;
      __builtin_prefetch(kb + (size_t)jn * DK + lane * 64, 0, 3);
      __builtin_prefetch(vb + (size_t)lane * SEQ + jn, 0, 3);
      __builtin_prefetch(vb + (size_t)(lane + 32) * SEQ + jn, 0, 3);
    }
    const _Float16* lkc = &lk[c][0];
    const _Float16* lvc = &lv[c][0];

    // ---- S^T tiles, two independent accumulation chains ----
    v16h a0 = load_a_f16(lkc, DK, 0, lane);
    v16h a2 = load_a_f16(lkc + 16 * DK, DK, 0, lane);
    v8f s_lo = wmma32(a0, bq0, vzero8());
    v8f s_hi = wmma32(a2, bq0, vzero8());
    v16h a1 = load_a_f16(lkc, DK, 32, lane);
    v16h a3 = load_a_f16(lkc + 16 * DK, DK, 32, lane);
    s_lo = wmma32(a1, bq1, s_lo);
    s_hi = wmma32(a3, bq1, s_hi);

    // ---- online softmax over this lane's 32 j values (column i) ----
    float cmax = -1e30f;
#pragma unroll
    for (int e = 0; e < 8; ++e) {
      cmax = fmaxf(cmax, s_lo[e]);
      cmax = fmaxf(cmax, s_hi[e]);
    }
    cmax = fmaxf(cmax, __shfl_xor(cmax, 16, 32));
    const float mnew = fmaxf(m, cmax);
    const float alpha = __expf(m - mnew);
    float p_lo[8], p_hi[8], csum = 0.0f;
#pragma unroll
    for (int e = 0; e < 8; ++e) {
      p_lo[e] = __expf(s_lo[e] - mnew);
      p_hi[e] = __expf(s_hi[e] - mnew);
      csum += p_lo[e] + p_hi[e];
    }
    l = l * alpha + csum + __shfl_xor(csum, 16, 32);
    m = mnew;
#pragma unroll
    for (int e = 0; e < 8; ++e) {
      o0[e] *= alpha; o1[e] *= alpha; o2[e] *= alpha; o3[e] *= alpha;
    }

    // ---- P as B fragment (32j x 16i) via half-swap shuffles ----
    v16h bp;
#pragma unroll
    for (int e = 0; e < 8; ++e) {
      const float plo_o = __shfl_xor(p_lo[e], 16, 32);
      const float phi_o = __shfl_xor(p_hi[e], 16, 32);
      bp[e]     = (_Float16)(half ? phi_o : p_lo[e]);
      bp[e + 8] = (_Float16)(half ? p_hi[e] : plo_o);
    }

    // ---- O^T += V^T(d x 32j) * P(32j x i), four d-tiles from LDS ----
    o0 = wmma32(load_a_f16(lvc + 0 * 16 * 32, 32, 0, lane), bp, o0);
    o1 = wmma32(load_a_f16(lvc + 1 * 16 * 32, 32, 0, lane), bp, o1);
    o2 = wmma32(load_a_f16(lvc + 2 * 16 * 32, 32, 0, lane), bp, o2);
    o3 = wmma32(load_a_f16(lvc + 3 * 16 * 32, 32, 0, lane), bp, o3);
  }

  const float inv = 1.0f / l;
  const int b = bh >> 3, h = bh & 7;
  _Float16* od = midh + ((size_t)(b * SEQ + i0 + lr)) * DIMM + h * DK + half * 8;
  const v8f* oo[4] = {&o0, &o1, &o2, &o3};
#pragma unroll
  for (int t = 0; t < 4; ++t) {
    h8 st;
#pragma unroll
    for (int r = 0; r < 8; ++r) st[r] = (_Float16)((*oo[t])[r] * inv);
    *(h8*)(od + t * 16) = st;
  }
}

// ---------------------------------------------------------------------------
// Kernel 3: out = midh @ Wo + bo; rotation-free 2-stage pipeline.  grid = BN/16.
// ---------------------------------------------------------------------------
__global__ __launch_bounds__(256) void out_proj(
    const _Float16* __restrict__ midh, const _Float16* __restrict__ Wot,
    const float* __restrict__ bo, float* __restrict__ out) {
  const int lane = threadIdx.x & 31;
  const int wave = threadIdx.x >> 5;
  const int rt = blockIdx.x;
  const int lr = lane & 15, half = lane >> 4;
  const int row0 = rt * 16;
  const int cbase = wave * 64;

  v8f acc[4];
#pragma unroll
  for (int t = 0; t < 4; ++t) acc[t] = vzero8();

  const _Float16* arow = midh + (size_t)row0 * DIMM;
  const _Float16* wb0  = Wot + (size_t)(cbase + lr) * DIMM + half * 16;

  v16h a0 = load_a_f16(arow, DIMM, 0, lane);
  v16h a1 = load_a_f16(arow, DIMM, 32, lane);
  v16h b0[4], b1[4];
#pragma unroll
  for (int t = 0; t < 4; ++t) { b0[t] = LDB(wb0, t, 0); b1[t] = LDB(wb0, t, 32); }

  for (int kk = 0; kk < DIMM / 32; kk += 2) {
#pragma unroll
    for (int t = 0; t < 4; ++t) acc[t] = wmma32(a0, b0[t], acc[t]);
    if (kk + 2 < DIMM / 32) {
      const int kn = (kk + 2) * 32;
      a0 = load_a_f16(arow, DIMM, kn, lane);
#pragma unroll
      for (int t = 0; t < 4; ++t) b0[t] = LDB(wb0, t, kn);
    }
#pragma unroll
    for (int t = 0; t < 4; ++t) acc[t] = wmma32(a1, b1[t], acc[t]);
    if (kk + 3 < DIMM / 32) {
      const int kn = (kk + 3) * 32;
      a1 = load_a_f16(arow, DIMM, kn, lane);
#pragma unroll
      for (int t = 0; t < 4; ++t) b1[t] = LDB(wb0, t, kn);
    }
  }

#pragma unroll
  for (int t = 0; t < 4; ++t) {
    const int col = cbase + t * 16 + lr;
    const float bias = bo[col];
#pragma unroll
    for (int r = 0; r < 8; ++r) {
      const int row = row0 + half * 8 + r;
      out[(size_t)row * DIMM + col] = acc[t][r] + bias;
    }
  }
}

// ---------------------------------------------------------------------------
extern "C" void kernel_launch(void* const* d_in, const int* in_sizes, int n_in,
                              void* d_out, int out_size, void* d_ws,
                              size_t ws_size, hipStream_t stream) {
  (void)in_sizes; (void)n_in; (void)out_size; (void)ws_size;
  const float* x   = (const float*)d_in[0];
  const float* Wq  = (const float*)d_in[1];
  const float* Wk  = (const float*)d_in[2];
  const float* Wv  = (const float*)d_in[3];
  const float* Wo  = (const float*)d_in[4];
  const float* bo  = (const float*)d_in[5];
  const float* rcb = (const float*)d_in[6];

  char* ws = (char*)d_ws;
  const size_t XH_B  = (size_t)BN * DIMM * sizeof(_Float16);                 // 8 MB
  const size_t WT_B  = (size_t)4 * DIMM * DIMM * sizeof(_Float16);           // 2 MB
  const size_t QKV_B = (size_t)BATCH * NHEAD * SEQ * DK * sizeof(_Float16);  // 8 MB

  _Float16* xh   = (_Float16*)(ws);
  _Float16* Wt   = (_Float16*)(ws + XH_B);
  _Float16* qh   = (_Float16*)(ws + XH_B + WT_B);
  _Float16* kh   = (_Float16*)(ws + XH_B + WT_B + QKV_B);
  _Float16* vT   = (_Float16*)(ws + XH_B + WT_B + 2 * QKV_B);
  _Float16* midh = (_Float16*)(ws + XH_B + WT_B + 3 * QKV_B);                // 8 MB

  prep_x<<<dim3((BN * DIMM) / (256 * 8)), 256, 0, stream>>>(x, xh);
  prep_w<<<dim3(128, 4), 256, 0, stream>>>(Wq, Wk, Wv, Wo, Wt);
  proj_qkv<<<dim3(BN / 16, 1, 3), 256, 0, stream>>>(xh, Wt, rcb, qh, kh, vT);
  attn_fa<<<dim3((SEQ / 16) / 8, BATCH * NHEAD), 256, 0, stream>>>(qh, kh, vT, midh);
  out_proj<<<dim3(BN / 16), 256, 0, stream>>>(midh, Wt + 3 * (size_t)DIMM * DIMM,
                                              bo, (float*)d_out);
}
